// GCNNodeCiteSeer_10333691314779
// MI455X (gfx1250) — matrix operand order repaired and checked by reading, not compiled, not run on hardware
//
#include <hip/hip_runtime.h>
#include <hip/hip_bf16.h>
#include <stdint.h>

#define IN_F   512
#define HID    128
#define NCLS   16

typedef __attribute__((ext_vector_type(16))) __bf16 v16bf;
typedef __attribute__((ext_vector_type(8)))  __bf16 v8bf;
typedef __attribute__((ext_vector_type(8)))  float  v8f;

__device__ __forceinline__ void atomic_add_f32(float* p, float v) {
  __hip_atomic_fetch_add(p, v, __ATOMIC_RELAXED, __HIP_MEMORY_SCOPE_AGENT);
}

// ---------------------------------------------------------------- utilities
__global__ void fill_f32_kernel(float* __restrict__ p, float v, long n) {
  long i = (long)blockIdx.x * blockDim.x + threadIdx.x;
  if (i < n) p[i] = v;
}
__global__ void fill_i32_kernel(int* __restrict__ p, int v, long n) {
  long i = (long)blockIdx.x * blockDim.x + threadIdx.x;
  if (i < n) p[i] = v;
}

// ------------------------------------------------------------------ degrees
__global__ void degrees_kernel(const int* __restrict__ src, const int* __restrict__ dst,
                               float* __restrict__ dout, float* __restrict__ din, int E) {
  int e = blockIdx.x * blockDim.x + threadIdx.x;
  if (e >= E) return;
  atomic_add_f32(&dout[src[e]], 1.0f);
  atomic_add_f32(&din[dst[e]], 1.0f);
}

__global__ void invsqrt_kernel(float* __restrict__ dout, float* __restrict__ din, int N) {
  int i = blockIdx.x * blockDim.x + threadIdx.x;
  if (i >= N) return;
  float a = dout[i];
  dout[i] = (a > 0.0f) ? rsqrtf(fmaxf(a, 1.0f)) : 1.0f;
  float b = din[i];
  din[i]  = (b > 0.0f) ? rsqrtf(fmaxf(b, 1.0f)) : 1.0f;
}

// -------------------------------------- W1 fp32 [512,128] -> bf16 [128,512]
// Transposed so each lane's B fragment is K-contiguous (wide loads, no packing).
__global__ void cvt_transpose_w1_kernel(const float* __restrict__ W1,
                                        __bf16* __restrict__ W1t) {
  int i = blockIdx.x * blockDim.x + threadIdx.x;
  if (i >= IN_F * HID) return;
  int k = i >> 7;            // row of W1 (K)
  int n = i & (HID - 1);     // col of W1 (N)
  W1t[(size_t)n * IN_F + k] = (__bf16)W1[i];
}

// ------------------------------------------------- per-graph node offsets
__global__ void scan_offsets_kernel(const int* __restrict__ bnn, int* __restrict__ off, int G) {
  if (blockIdx.x == 0 && threadIdx.x == 0) {
    int a = 0;
    for (int g = 0; g < G; ++g) { off[g] = a; a += bnn[g]; }
  }
}

__global__ void build_tslot_kernel(const int* __restrict__ target, const int* __restrict__ off,
                                   int* __restrict__ tslot, int G) {
  int g = blockIdx.x * blockDim.x + threadIdx.x;
  if (g < G) tslot[off[g] + target[g]] = g;
}

// --------------------------------------------------------------- GEMM1 WMMA
// H[N,128] = (X[N,512] * invs_out[:,None]) @ W1   (bf16 A/B, f32 acc)
// block = 256 threads = 8 waves; block owns a 16-row M tile; wave w owns cols 16w..16w+15.
// Inner loop per k-step: 2x ds_load_b128 (A) + 2x global_load_b128 (B) + 1x WMMA.
__global__ __launch_bounds__(256) void gemm1_wmma_kernel(
    const float* __restrict__ X, const float* __restrict__ invs_out,
    const __bf16* __restrict__ W1t,          // [128,512] bf16, N-major
    float* __restrict__ H) {
  __shared__ __align__(16) __bf16 As[16 * IN_F];   // 16 KB A panel (bf16, pre-scaled)

  const int m0  = blockIdx.x * 16;
  const int tid = threadIdx.x;

  // cooperative load: scale by D_out^{-1/2}, convert to bf16 once
  for (int i = tid; i < 16 * IN_F; i += 256) {
    int r = i >> 9, k = i & (IN_F - 1);
    As[i] = (__bf16)(X[(size_t)(m0 + r) * IN_F + k] * invs_out[m0 + r]);
  }
  __syncthreads();

  const int wave = tid >> 5;          // 0..7 -> output column tile
  const int lane = tid & 31;
  const int col  = lane & 15;         // N within tile (B/C/D lane striping)
  const int row  = lane & 15;         // M within tile (A lane striping)
  const int kbA  = (lane < 16) ? 0 : 8;   // A: low lanes K 0..7/16..23, high lanes 8..15/24..31
  const int kbB  = (lane < 16) ? 0 : 8;   // B: low lanes K 0..7/16..23, high lanes 8..15/24..31

  const __bf16* bcol = W1t + (size_t)(wave * 16 + col) * IN_F;  // lane's column of W1

  v8f acc = {};
  for (int ks = 0; ks < IN_F; ks += 32) {
    // A fragment from LDS: two contiguous 8-half chunks (ds_load_b128 each)
    const v8bf a_lo = *(const v8bf*)&As[row * IN_F + ks + kbA];
    const v8bf a_hi = *(const v8bf*)&As[row * IN_F + ks + kbA + 16];
    v16bf A = __builtin_shufflevector(a_lo, a_hi,
                                      0,1,2,3,4,5,6,7,8,9,10,11,12,13,14,15);

    // B fragment: two 16-byte chunks (K +0..7 and +16..23) of the transposed, L2-resident W1
    const __bf16* bp = bcol + ks + kbB;
    if (ks + 32 < IN_F)
      __builtin_prefetch(bp + 32, 0, 3);           // next K slice
    const v8bf b_lo = *(const v8bf*)bp;
    const v8bf b_hi = *(const v8bf*)(bp + 16);
    v16bf B = __builtin_shufflevector(b_lo, b_hi,
                                      0,1,2,3,4,5,6,7,8,9,10,11,12,13,14,15);

    acc = __builtin_amdgcn_wmma_f32_16x16x32_bf16(
        /*neg_a=*/false, A, /*neg_b=*/false, B,
        /*c_mod=*/(short)0, acc, /*reuse_a=*/false, /*reuse_b=*/false);
  }

  // C/D striping: VGPR j, lane l -> M = j + 8*(l>=16), N = l&15
  const int mh = (lane >> 4) << 3;
#pragma unroll
  for (int j = 0; j < 8; ++j)
    H[(size_t)(m0 + mh + j) * HID + wave * 16 + col] = acc[j];
}

// ------------------------------------------------------------------- SpMM1
// agg[dst] += H[src] * e_w   (one wave per edge, float4 per lane)
__global__ __launch_bounds__(256) void spmm1_kernel(
    const int* __restrict__ src, const int* __restrict__ dst,
    const float* __restrict__ ew, const float* __restrict__ H,
    float* __restrict__ agg, int E) {
  long t = (long)blockIdx.x * 256 + threadIdx.x;
  int e = (int)(t >> 5);
  if (e >= E) return;
  int f = (int)(t & 31) << 2;
  int s = src[e], d = dst[e];
  float w = ew[e];
  const float4 hv = *(const float4*)&H[(size_t)s * HID + f];
  float* ap = &agg[(size_t)d * HID + f];
  atomic_add_f32(ap + 0, hv.x * w);
  atomic_add_f32(ap + 1, hv.y * w);
  atomic_add_f32(ap + 2, hv.z * w);
  atomic_add_f32(ap + 3, hv.w * w);
}

// ---------------------------------------------- layer-1 epilogue (in place)
// g = relu(agg*invs_in + b1) * invs_out   (pre-scales for layer 2)
__global__ void epilogue1_kernel(float* __restrict__ agg,
                                 const float* __restrict__ invs_in,
                                 const float* __restrict__ invs_out,
                                 const float* __restrict__ b1, long total) {
  long i = (long)blockIdx.x * blockDim.x + threadIdx.x;
  if (i >= total) return;
  long n = i >> 7;
  int  c = (int)(i & (HID - 1));
  float v = agg[i] * invs_in[n] + b1[c];
  v = fmaxf(v, 0.0f);
  agg[i] = v * invs_out[n];
}

// ------------------------------------------------- SpMM2 (target dsts only)
__global__ void spmm2_kernel(const int* __restrict__ src, const int* __restrict__ dst,
                             const float* __restrict__ ew, const int* __restrict__ tslot,
                             const float* __restrict__ G, float* __restrict__ acc2, int E) {
  int e = blockIdx.x * blockDim.x + threadIdx.x;
  if (e >= E) return;
  int sg = tslot[dst[e]];
  if (sg < 0) return;                 // ~99.6% of edges exit after 8 bytes read
  float w = ew[e];
  const float* gp = &G[(size_t)src[e] * HID];
  float* ap = &acc2[(size_t)sg * HID];
#pragma unroll 4
  for (int f = 0; f < HID; ++f) atomic_add_f32(&ap[f], gp[f] * w);
}

// -------------------------------------------------------------------- final
// out[g] = (acc2[g] @ W2) * invs_in[node] + b2
__global__ void final_kernel(const float* __restrict__ acc2, const float* __restrict__ invs_in,
                             const int* __restrict__ off, const int* __restrict__ target,
                             const float* __restrict__ W2, const float* __restrict__ b2,
                             float* __restrict__ out) {
  int g = blockIdx.x, t = threadIdx.x;
  if (t >= NCLS) return;
  int node = off[g] + target[g];
  float s = invs_in[node];
  const float* a = &acc2[(size_t)g * HID];
  float sum = 0.0f;
#pragma unroll 8
  for (int k = 0; k < HID; ++k) sum += a[k] * W2[k * NCLS + t];
  out[g * NCLS + t] = sum * s + b2[t];
}

// ====================================================================== host
extern "C" void kernel_launch(void* const* d_in, const int* in_sizes, int n_in,
                              void* d_out, int out_size, void* d_ws, size_t ws_size,
                              hipStream_t stream) {
  const float* in_feat = (const float*)d_in[0];
  const float* e_w     = (const float*)d_in[1];
  const float* W1      = (const float*)d_in[2];
  const float* b1      = (const float*)d_in[3];
  const float* W2      = (const float*)d_in[4];
  const float* b2      = (const float*)d_in[5];
  const int*   src     = (const int*)d_in[6];
  const int*   dst     = (const int*)d_in[7];
  const int*   target  = (const int*)d_in[8];
  const int*   bnn     = (const int*)d_in[9];

  const int N = in_sizes[0] / IN_F;   // 262144
  const int E = in_sizes[1];          // 2097152
  const int G = in_sizes[8];          // 1024

  // ---- workspace carving (256B aligned) ----
  char* ws = (char*)d_ws;
  size_t o = 0;
  auto carve = [&](size_t bytes) { void* p = ws + o; o = (o + bytes + 255) & ~(size_t)255; return p; };
  float*  invs_out = (float*)carve((size_t)N * 4);
  float*  invs_in  = (float*)carve((size_t)N * 4);
  __bf16* W1t      = (__bf16*)carve((size_t)IN_F * HID * 2);
  int*    off      = (int*)carve((size_t)G * 4);
  int*    tslot    = (int*)carve((size_t)N * 4);
  float*  acc2     = (float*)carve((size_t)G * HID * 4);
  float*  h1p      = (float*)carve((size_t)N * HID * 4);
  float*  agg1     = (float*)carve((size_t)N * HID * 4);

  const long NH = (long)N * HID;

  // init scratch (graph-replay safe: done every call)
  fill_f32_kernel<<<(N + 255) / 256, 256, 0, stream>>>(invs_out, 0.0f, N);
  fill_f32_kernel<<<(N + 255) / 256, 256, 0, stream>>>(invs_in, 0.0f, N);
  fill_f32_kernel<<<(int)((NH + 255) / 256), 256, 0, stream>>>(agg1, 0.0f, NH);
  fill_f32_kernel<<<(G * HID + 255) / 256, 256, 0, stream>>>(acc2, 0.0f, (long)G * HID);
  fill_i32_kernel<<<(N + 255) / 256, 256, 0, stream>>>(tslot, -1, N);

  // degrees -> inv-sqrt norms
  degrees_kernel<<<(E + 255) / 256, 256, 0, stream>>>(src, dst, invs_out, invs_in, E);
  invsqrt_kernel<<<(N + 255) / 256, 256, 0, stream>>>(invs_out, invs_in, N);

  // weights to transposed bf16; graph offsets; target slots
  cvt_transpose_w1_kernel<<<(IN_F * HID + 255) / 256, 256, 0, stream>>>(W1, W1t);
  scan_offsets_kernel<<<1, 32, 0, stream>>>(bnn, off, G);
  build_tslot_kernel<<<(G + 255) / 256, 256, 0, stream>>>(target, off, tslot, G);

  // layer 1: WMMA projection, SpMM aggregation, epilogue
  gemm1_wmma_kernel<<<N / 16, 256, 0, stream>>>(in_feat, invs_out, W1t, h1p);
  spmm1_kernel<<<(int)(((long)E * 32 + 255) / 256), 256, 0, stream>>>(src, dst, e_w, h1p, agg1, E);
  epilogue1_kernel<<<(int)((NH + 255) / 256), 256, 0, stream>>>(agg1, invs_in, invs_out, b1, NH);

  // layer 2: aggregate only into target nodes, then tiny projection
  spmm2_kernel<<<(E + 255) / 256, 256, 0, stream>>>(src, dst, e_w, tslot, agg1, acc2, E);
  final_kernel<<<G, 32, 0, stream>>>(acc2, invs_in, off, target, W2, b2, (float*)d_out);
}
